// AttentionModule_ReLU_74741020885605
// MI455X (gfx1250) — matrix operand order repaired and checked by reading, not compiled, not run on hardware
//
#include <hip/hip_runtime.h>
#include <cstdint>

#define DEV __device__ __forceinline__

typedef __bf16 v8bf  __attribute__((ext_vector_type(8)));
typedef __bf16 v16bf __attribute__((ext_vector_type(16)));
typedef float  v4f   __attribute__((ext_vector_type(4)));
typedef float  v8f   __attribute__((ext_vector_type(8)));

constexpr int Bb = 8;
constexpr int Nn = 2048;
constexpr int Dd = 1024;
constexpr int Cc = 1024;
constexpr int Hh = 512;

DEV v8f vzero8() {
  v8f z;
#pragma unroll
  for (int i = 0; i < 8; i++) z[i] = 0.f;
  return z;
}

// A fragment from bf16 memory: this file's k order (16-bit A, 16x32; B uses the same):
// lane row M = lane&15, K contiguous {off..off+15},
// off = (lane>=16 ? 16 : 0). Caller pre-offsets pointer; chunks p[0..7], p[8..15].
DEV v16bf afrag(const __bf16* p) {
  v8bf c0 = *(const v8bf*)p;
  v8bf c1 = *(const v8bf*)(p + 8);
  v16bf r;
#pragma unroll
  for (int i = 0; i < 8; i++) {
    r[i]     = c0[i];
    r[8 + i] = c1[i];
  }
  return r;
}

DEV v8f wmma_bf16(v16bf a, v16bf b, v8f c) {
  return __builtin_amdgcn_wmma_f32_16x16x32_bf16(
      /*neg_a=*/false, a, /*neg_b=*/false, b,
      /*c_mod=*/(short)0, c, /*reuse_a=*/false, /*reuse_b=*/false);
}

// ---- CDNA5 async copy: global -> LDS, 16B granules, ASYNCcnt tracked ----
// GVS mode: mem = SADDR(sgpr pair) + VADDR(vgpr u32) ; LDS[vdst_u32].
// Generic pointers to LDS carry the LDS byte offset in their low 32 bits
// (aperture layout, ISA 10.2), so (unsigned)(uintptr_t) gives VDST.
DEV void async_copy_b128(unsigned lds_off, const void* gbase, unsigned goff) {
  asm volatile("global_load_async_to_lds_b128 %0, %1, %2"
               :
               : "v"(lds_off), "v"(goff), "s"(gbase)
               : "memory");
}
DEV void wait_async0() { asm volatile("s_wait_asynccnt 0" ::: "memory"); }

// ---------------------------------------------------------------------------
// fp32 -> bf16 elementwise (8 elems/thread); n is a multiple of 8.
// ---------------------------------------------------------------------------
__launch_bounds__(256)
__global__ void f32_to_bf16(const float* __restrict__ in,
                            __bf16* __restrict__ out, int n) {
  int i = (blockIdx.x * 256 + threadIdx.x) * 8;
  if (i + 8 <= n) {
    v4f a = *(const v4f*)(in + i);
    v4f b = *(const v4f*)(in + i + 4);
    v8bf o;
#pragma unroll
    for (int j = 0; j < 4; j++) {
      o[j]     = (__bf16)a[j];
      o[4 + j] = (__bf16)b[j];
    }
    *(v8bf*)(out + i) = o;
  }
}

// ---------------------------------------------------------------------------
// Fused projection: Y = ReLU(LayerNorm(X @ W^T + bias) * g + beta)
// X, W already bf16. One block = 16 rows x OUTW cols (full LN row in-block).
// A tile (16 x Dd bf16 = 32 KB, contiguous) staged in LDS via async copy.
// 8 waves; wave w owns TPW = OUTW/128 tiles of 16x16.
// OMODE: 0 = bf16 row-major [rows][OUTW]
//        1 = bf16 transposed per batch: out[b][col][n]  (for v1)
//        2 = fp32 into d_out with row stride Cc, col offset 0 (for v2)
// ---------------------------------------------------------------------------
template <int OUTW, int OMODE>
__launch_bounds__(256)
__global__ void proj_ln_relu(const __bf16* __restrict__ X,
                             const __bf16* __restrict__ W,
                             const float* __restrict__ bias,
                             const float* __restrict__ gam,
                             const float* __restrict__ beta,
                             void* __restrict__ outp) {
  constexpr int TPW = OUTW / 128;
  extern __shared__ char smemraw[];
  __bf16* As = (__bf16*)smemraw;                 // [16][Dd]  32 KB
  float* y   = (float*)(As + 16 * Dd);           // [16][OUTW]
  float* red = y + 16 * OUTW;                    // [16*16*2]
  float* mv  = red + 16 * 16 * 2;                // [16*2]

  const int tid = threadIdx.x;
  const int wave = tid >> 5, lane = tid & 31;
  const int lrow = lane & 15, hi = lane >> 4;
  const int m0 = blockIdx.x * 16;

  {  // async stage of the contiguous 32 KB A tile into LDS
    const unsigned ldsbase = (unsigned)(uintptr_t)As;
    const __bf16* g0 = X + (size_t)m0 * Dd;
    for (int i = tid; i < 16 * Dd * 2 / 16; i += 256)
      async_copy_b128(ldsbase + i * 16, (const void*)g0, (unsigned)(i * 16));
    wait_async0();
  }
  __syncthreads();

  v8f acc[TPW];
#pragma unroll
  for (int t = 0; t < TPW; t++) acc[t] = vzero8();

  const __bf16* aptr0 = As + lrow * Dd + hi * 16;
  const __bf16* brow0 = W + (size_t)(wave * (16 * TPW) + lrow) * Dd + hi * 16;
  for (int k0 = 0; k0 < Dd; k0 += 32) {
    v16bf a = afrag(aptr0 + k0);
#pragma unroll
    for (int t = 0; t < TPW; t++) {
      const __bf16* brow = brow0 + (size_t)(t * 16) * Dd + k0;
      v16bf b = *(const v16bf*)brow;
      if (k0 + 32 < Dd) __builtin_prefetch(brow + 32, 0, 0);
      acc[t] = wmma_bf16(a, b, acc[t]);
    }
  }

  // C/D layout: lane covers col N=lane&15; VGPR r holds row M=r+8*hi.
#pragma unroll
  for (int t = 0; t < TPW; t++) {
    const int col = wave * (16 * TPW) + t * 16 + lrow;
    const float bi = bias[col];
#pragma unroll
    for (int r = 0; r < 8; r++)
      y[(r + hi * 8) * OUTW + col] = acc[t][r] + bi;
  }
  __syncthreads();

  // LayerNorm over OUTW per row: 16 threads per row, interleaved columns.
  const int row = tid >> 4, seg = tid & 15;
  {
    float s = 0.f, ss = 0.f;
    for (int j = 0; j < OUTW / 16; j++) {
      float v = y[row * OUTW + j * 16 + seg];
      s += v;
      ss += v * v;
    }
    red[(row * 16 + seg) * 2] = s;
    red[(row * 16 + seg) * 2 + 1] = ss;
  }
  __syncthreads();
  if (seg == 0) {
    float s = 0.f, ss = 0.f;
    for (int j = 0; j < 16; j++) {
      s += red[(row * 16 + j) * 2];
      ss += red[(row * 16 + j) * 2 + 1];
    }
    const float mu = s / (float)OUTW;
    const float var = ss / (float)OUTW - mu * mu;
    mv[row * 2] = mu;
    mv[row * 2 + 1] = rsqrtf(var + 1e-5f);
  }
  __syncthreads();
  const float mu = mv[row * 2], rstd = mv[row * 2 + 1];
  const int gm = m0 + row;
  for (int j = 0; j < OUTW / 16; j++) {
    const int col = j * 16 + seg;
    float v = (y[row * OUTW + col] - mu) * rstd * gam[col] + beta[col];
    v = fmaxf(v, 0.f);
    if constexpr (OMODE == 0) {
      ((__bf16*)outp)[(size_t)gm * OUTW + col] = (__bf16)v;
    } else if constexpr (OMODE == 1) {
      const int bidx = gm / Nn, n = gm % Nn;
      ((__bf16*)outp)[((size_t)bidx * OUTW + col) * Nn + n] = (__bf16)v;
    } else {
      ((float*)outp)[(size_t)gm * Cc + col] = v;
    }
  }
}

// ---------------------------------------------------------------------------
// Attention: one block = (batch b, 16 query rows).
//   1) async-stage q tile [16][Cc] bf16 in LDS
//   2) scores = q @ k^T : wave w covers keys [w*256, w*256+256), bf16 -> LDS
//   3) block-cooperative row softmax (fp32 math, bf16 storage, in place)
//   4) out = P @ v1 : wave w covers H cols [w*64, w*64+64), fp32 to d_out
// v1 stored transposed [B][H][N] so B-fragment loads are contiguous.
// ---------------------------------------------------------------------------
__launch_bounds__(256)
__global__ void attn_softmax_pv(const __bf16* __restrict__ Q,
                                const __bf16* __restrict__ Kc,
                                const __bf16* __restrict__ V1T,
                                float* __restrict__ Out) {
  extern __shared__ char smemraw[];
  __bf16* qs = (__bf16*)smemraw;         // [16][Cc]  32 KB
  __bf16* sc = qs + 16 * Cc;             // [16][Nn]  64 KB
  float* red = (float*)(sc + 16 * Nn);   // [256]
  float* rstat = red + 256;              // [16]

  const int tid = threadIdx.x;
  const int wave = tid >> 5, lane = tid & 31;
  const int lrow = lane & 15, hi = lane >> 4;
  const int b = blockIdx.x >> 7;             // Nn/16 = 128 row-tiles per batch
  const int q0 = (blockIdx.x & 127) * 16;
  const size_t qrow0 = (size_t)b * Nn + q0;

  {  // async stage of contiguous 32 KB q tile into LDS
    const unsigned ldsbase = (unsigned)(uintptr_t)qs;
    const __bf16* g0 = Q + qrow0 * Cc;
    for (int i = tid; i < 16 * Cc * 2 / 16; i += 256)
      async_copy_b128(ldsbase + i * 16, (const void*)g0, (unsigned)(i * 16));
    wait_async0();
  }
  __syncthreads();

  // -------- scores --------
  for (int kt = 0; kt < 16; kt++) {
    const int key0 = wave * 256 + kt * 16;
    v8f acc = vzero8();
    const __bf16* aptr0 = qs + lrow * Cc + hi * 16;
    const __bf16* brow0 = Kc + ((size_t)b * Nn + key0 + lrow) * Cc + hi * 16;
    for (int k0 = 0; k0 < Cc; k0 += 32) {
      v16bf a = afrag(aptr0 + k0);
      v16bf bb = *(const v16bf*)(brow0 + k0);
      acc = wmma_bf16(a, bb, acc);
    }
#pragma unroll
    for (int r = 0; r < 8; r++)
      sc[(r + hi * 8) * Nn + key0 + lrow] = (__bf16)acc[r];
  }
  __syncthreads();

  // -------- softmax (rows of length Nn) --------
  const int row = tid >> 4, seg = tid & 15;
  {
    float m = -3.0e38f;
    for (int j = 0; j < Nn / 16; j++)
      m = fmaxf(m, (float)sc[row * Nn + j * 16 + seg]);
    red[row * 16 + seg] = m;
  }
  __syncthreads();
  if (seg == 0) {
    float m = red[row * 16];
    for (int j = 1; j < 16; j++) m = fmaxf(m, red[row * 16 + j]);
    rstat[row] = m;
  }
  __syncthreads();
  {
    const float m = rstat[row];
    float s = 0.f;
    for (int j = 0; j < Nn / 16; j++) {
      const int idx = row * Nn + j * 16 + seg;
      const float e = __expf((float)sc[idx] - m);
      s += e;
      sc[idx] = (__bf16)e;
    }
    red[row * 16 + seg] = s;
  }
  __syncthreads();
  if (seg == 0) {
    float s = 0.f;
    for (int j = 0; j < 16; j++) s += red[row * 16 + j];
    rstat[row] = 1.f / s;
  }
  __syncthreads();
  {
    const float inv = rstat[row];
    for (int j = 0; j < Nn / 16; j++) {
      const int idx = row * Nn + j * 16 + seg;
      sc[idx] = (__bf16)((float)sc[idx] * inv);
    }
  }
  __syncthreads();

  // -------- P @ V1 --------
  v8f oacc[4];
#pragma unroll
  for (int t = 0; t < 4; t++) oacc[t] = vzero8();
  const int h0 = wave * 64;
  const __bf16* aptr0 = sc + lrow * Nn + hi * 16;
  for (int m0k = 0; m0k < Nn; m0k += 32) {
    v16bf a = afrag(aptr0 + m0k);
#pragma unroll
    for (int t = 0; t < 4; t++) {
      const __bf16* brow =
          V1T + ((size_t)b * Hh + h0 + t * 16 + lrow) * Nn + m0k + hi * 16;
      v16bf bb = *(const v16bf*)brow;
      oacc[t] = wmma_bf16(a, bb, oacc[t]);
    }
  }
#pragma unroll
  for (int t = 0; t < 4; t++) {
#pragma unroll
    for (int r = 0; r < 8; r++) {
      Out[(qrow0 + r + hi * 8) * Cc + Hh + h0 + t * 16 + lrow] = oacc[t][r];
    }
  }
}

// ---------------------------------------------------------------------------
extern "C" void kernel_launch(void* const* d_in, const int* in_sizes, int n_in,
                              void* d_out, int out_size, void* d_ws,
                              size_t ws_size, hipStream_t stream) {
  const float* sgm   = (const float*)d_in[0];
  const float* velo  = (const float*)d_in[1];
  const float* Wq    = (const float*)d_in[2];
  const float* bq    = (const float*)d_in[3];
  const float* gq    = (const float*)d_in[4];
  const float* betaq = (const float*)d_in[5];
  const float* Wk    = (const float*)d_in[6];
  const float* bk    = (const float*)d_in[7];
  const float* gk    = (const float*)d_in[8];
  const float* betak = (const float*)d_in[9];
  const float* Wv1    = (const float*)d_in[10];
  const float* bv1    = (const float*)d_in[11];
  const float* gv1    = (const float*)d_in[12];
  const float* betav1 = (const float*)d_in[13];
  const float* Wv2    = (const float*)d_in[14];
  const float* bv2    = (const float*)d_in[15];
  const float* gv2    = (const float*)d_in[16];
  const float* betav2 = (const float*)d_in[17];
  float* out = (float*)d_out;

  // Workspace (bf16): xb | vb | Wq | Wk | Wv1 | Wv2 | q | k | v1^T
  __bf16* xb   = (__bf16*)d_ws;
  __bf16* vb   = xb + (size_t)Bb * Nn * Dd;
  __bf16* wqb  = vb + (size_t)Bb * Nn * Dd;
  __bf16* wkb  = wqb + (size_t)Cc * Dd;
  __bf16* wv1b = wkb + (size_t)Cc * Dd;
  __bf16* wv2b = wv1b + (size_t)Hh * Dd;
  __bf16* qws  = wv2b + (size_t)Hh * Dd;
  __bf16* kws  = qws + (size_t)Bb * Nn * Cc;
  __bf16* v1t  = kws + (size_t)Bb * Nn * Cc;

  auto cvt = [&](const float* src, __bf16* dst, size_t n) {
    f32_to_bf16<<<(int)(n / (256 * 8)), 256, 0, stream>>>(src, dst, (int)n);
  };
  cvt(sgm, xb, (size_t)Bb * Nn * Dd);
  cvt(velo, vb, (size_t)Bb * Nn * Dd);
  cvt(Wq, wqb, (size_t)Cc * Dd);
  cvt(Wk, wkb, (size_t)Cc * Dd);
  cvt(Wv1, wv1b, (size_t)Hh * Dd);
  cvt(Wv2, wv2b, (size_t)Hh * Dd);

  const int blocks = Bb * Nn / 16;  // 1024
  const size_t shP1024 =
      (size_t)16 * Dd * 2 + (size_t)(16 * 1024 + 16 * 16 * 2 + 32) * 4;
  const size_t shP512 =
      (size_t)16 * Dd * 2 + (size_t)(16 * 512 + 16 * 16 * 2 + 32) * 4;

  proj_ln_relu<1024, 0><<<blocks, 256, shP1024, stream>>>(
      xb, wqb, bq, gq, betaq, (void*)qws);
  proj_ln_relu<1024, 0><<<blocks, 256, shP1024, stream>>>(
      vb, wkb, bk, gk, betak, (void*)kws);
  proj_ln_relu<512, 1><<<blocks, 256, shP512, stream>>>(
      vb, wv1b, bv1, gv1, betav1, (void*)v1t);
  proj_ln_relu<512, 2><<<blocks, 256, shP512, stream>>>(
      xb, wv2b, bv2, gv2, betav2, (void*)out);

  const size_t shA =
      (size_t)16 * Cc * 2 + (size_t)16 * Nn * 2 + (256 + 16) * 4;
  attn_softmax_pv<<<blocks, 256, shA, stream>>>(qws, kws, v1t, out);
}
